// IMVTensorMultiStepLSTM_Dropout_37065567764828
// MI455X (gfx1250) — compile-verified, matrix operand order checked by reading, not run on hardware
//
#include <hip/hip_runtime.h>
#include <cstdint>

typedef __attribute__((ext_vector_type(16))) _Float16 v16h;
typedef __attribute__((ext_vector_type(8)))  float    v8f;

#define B_      256
#define T_      96
#define D_      32
#define H_      128
#define OUT_    32
#define FSTEPS_ 24
#define ROWS_   32      // batch rows per cell workgroup
#define WS_     136     // padded half-stride for f16 LDS rows (16B aligned, bank-spread)
#define GS_     132     // padded float-stride for f32 LDS rows

__device__ __forceinline__ float sigmoidf_(float v){ return 1.0f/(1.0f + __expf(-v)); }

// ---------------------------------------------------------------------------
// Encoder: one workgroup owns (variable d, 32 batch rows) for the whole T scan.
// LDS-resident: f16 weights (transposed), f16 h, f32 c, attention accumulators.
// Emits v_wmma_f32_16x16x32_f16 for the 4 recurrent gate matmuls.
// ---------------------------------------------------------------------------
__global__ __launch_bounds__(256, 1) void enc_kernel(
    const float* __restrict__ x,
    const float* __restrict__ Uj, const float* __restrict__ Ui,
    const float* __restrict__ Uf, const float* __restrict__ Uo,
    const float* __restrict__ Wj, const float* __restrict__ Wi,
    const float* __restrict__ Wf, const float* __restrict__ Wo,
    const float* __restrict__ bj, const float* __restrict__ bi,
    const float* __restrict__ bf, const float* __restrict__ bo,
    const float* __restrict__ Fa, const float* __restrict__ Fab,
    float* __restrict__ out_alphas,    // [B,T,D]
    float* __restrict__ out_outputs,   // [B,T,D,H]
    float* __restrict__ ws_h, float* __restrict__ ws_c, float* __restrict__ ws_g)
{
    __shared__ _Float16 sW[4*H_*WS_];     // W^T per gate: [n][k], f16
    __shared__ _Float16 sH[ROWS_*WS_];    // current h, f16 (WMMA A operand)
    __shared__ float    sC[ROWS_*GS_];
    __shared__ float    sG[4*ROWS_*GS_];  // gate pre-activations
    __shared__ float    sNum[ROWS_*GS_];  // attention numerator  sum_t a_t*h_t
    __shared__ float    sA[ROWS_*T_];     // raw attention coefficients a_t
    __shared__ float    sU[4*H_], sBias[4*H_], sFa[H_], sX[ROWS_], sDen[ROWS_];

    const int tid  = threadIdx.x;
    const int wave = tid >> 5, lane = tid & 31;
    const int d    = blockIdx.x & (D_-1);
    const int row0 = (blockIdx.x >> 5) * ROWS_;

    const float* Wsrc[4] = {Wj, Wi, Wf, Wo};
    const float* Usrc[4] = {Uj, Ui, Uf, Uo};
    const float* bsrc[4] = {bj, bi, bf, bo};

    // Load weights transposed into LDS as f16 (coalesced global reads).
    for (int g = 0; g < 4; ++g){
        const float* Wg = Wsrc[g] + (size_t)d*H_*H_;
        for (int idx = tid; idx < H_*H_; idx += 256){
            int k = idx >> 7, n = idx & (H_-1);
            sW[g*H_*WS_ + n*WS_ + k] = (_Float16)Wg[idx];
        }
    }
    for (int idx = tid; idx < 4*H_; idx += 256){
        int g = idx >> 7, col = idx & (H_-1);
        sU[idx]    = Usrc[g][d*H_ + col];
        sBias[idx] = bsrc[g][d*H_ + col];
    }
    if (tid < H_) sFa[tid] = Fa[d*H_ + tid];
    const float fab = Fab[d];
    for (int e = tid; e < ROWS_*H_; e += 256){
        int r = e >> 7, c = e & (H_-1);
        sH[r*WS_ + c] = (_Float16)0.f;
        sC[r*GS_ + c] = 0.f;
        sNum[r*GS_ + c] = 0.f;
    }
    if (tid < ROWS_) sDen[tid] = 0.f;
    __syncthreads();

    for (int t = 0; t < T_; ++t){
        if (tid < ROWS_) sX[tid] = x[(size_t)(row0+tid)*T_*D_ + t*D_ + d];

        // 64 output tiles (4 gates x 2 mt x 8 nt), 8 per wave, K = 4 x 32.
        #pragma unroll
        for (int q = 0; q < 8; ++q){
            const int tIdx = wave*8 + q;
            const int g = tIdx >> 4, rem = tIdx & 15, mt = rem >> 3, nt = rem & 7;
            const int hs8  = (lane >> 4) << 3;       // K-half select per ISA layout
            const int arow = mt*16 + (lane & 15);
            const int bcol = nt*16 + (lane & 15);
            v8f acc = {};
            #pragma unroll
            for (int kb = 0; kb < 4; ++kb){
                const _Float16* ap = &sH[arow*WS_ + kb*32 + hs8];
                const _Float16* bp = &sW[g*H_*WS_ + bcol*WS_ + kb*32 + hs8];
                v16h a, b;
                #pragma unroll
                for (int i = 0; i < 8; ++i){
                    a[i] = ap[i];  a[i+8] = ap[16+i];
                    b[i] = bp[i];  b[i+8] = bp[16+i];
                }
                acc = __builtin_amdgcn_wmma_f32_16x16x32_f16(
                          false, a, false, b, (short)0, acc, false, false);
            }
            const int cm0 = mt*16 + hs8;             // C/D layout: M = vgpr + 8*(lane>=16)
            #pragma unroll
            for (int v = 0; v < 8; ++v)
                sG[(g*ROWS_ + cm0 + v)*GS_ + bcol] = acc[v];
        }
        __syncthreads();

        // Fused elementwise LSTM update + outputs store.
        for (int e = tid; e < ROWS_*H_; e += 256){
            int r = e >> 7, c = e & (H_-1);
            float xr = sX[r];
            float jv = tanhf    (sG[(0*ROWS_+r)*GS_+c] + xr*sU[0*H_+c] + sBias[0*H_+c]);
            float iv = sigmoidf_(sG[(1*ROWS_+r)*GS_+c] + xr*sU[1*H_+c] + sBias[1*H_+c]);
            float fv = sigmoidf_(sG[(2*ROWS_+r)*GS_+c] + xr*sU[2*H_+c] + sBias[2*H_+c]);
            float ov = sigmoidf_(sG[(3*ROWS_+r)*GS_+c] + xr*sU[3*H_+c] + sBias[3*H_+c]);
            float c2 = sC[r*GS_+c]*fv + iv*jv;
            float h2 = ov*tanhf(c2);
            sC[r*GS_+c] = c2;
            sH[r*WS_+c] = (_Float16)h2;
            out_outputs[(((size_t)(row0+r)*T_ + t)*D_ + d)*H_ + c] = h2;
        }
        __syncthreads();

        // Streaming attention: a_t = exp(tanh(h . Fa + b)); accumulate num/den.
        #pragma unroll
        for (int rr = 0; rr < 4; ++rr){
            int r = wave + rr*8;
            float p = 0.f;
            #pragma unroll
            for (int c = lane; c < H_; c += 32) p += (float)sH[r*WS_+c]*sFa[c];
            #pragma unroll
            for (int off = 16; off > 0; off >>= 1) p += __shfl_xor(p, off);
            float av = __expf(tanhf(p + fab));
            if (lane == 0){ sA[r*T_ + t] = av; sDen[r] += av; }
            #pragma unroll
            for (int c = lane; c < H_; c += 32) sNum[r*GS_+c] += av*(float)sH[r*WS_+c];
        }
        __syncthreads();
    }

    // Finalize: g_n, hT, cT to workspace; normalized alphas to output.
    for (int e = tid; e < ROWS_*H_; e += 256){
        int r = e >> 7, c = e & (H_-1);
        size_t o = ((size_t)(row0+r)*D_ + d)*H_ + c;
        ws_g[o] = sNum[r*GS_+c] / sDen[r];
        ws_h[o] = (float)sH[r*WS_+c];
        ws_c[o] = sC[r*GS_+c];
    }
    for (int e = tid; e < ROWS_*T_; e += 256){
        int r = e / T_, tt = e % T_;
        out_alphas[((size_t)(row0+r)*T_ + tt)*D_ + d] = sA[r*T_+tt] / sDen[r];
    }
}

// ---------------------------------------------------------------------------
// Forecast head: mu/beta softmax over variables, y, betas, next input `prev`.
// One block per batch row; small VALU-bound kernel.
// ---------------------------------------------------------------------------
__global__ __launch_bounds__(256, 1) void fhead_kernel(
    const float* __restrict__ ws_g, const float* __restrict__ ws_h,
    const float* __restrict__ Fbw, const float* __restrict__ Fbb,
    const float* __restrict__ Phw, const float* __restrict__ Phb,
    const float* __restrict__ Pw,  const float* __restrict__ Pb,
    float* __restrict__ out_fore, float* __restrict__ out_betas,
    float* __restrict__ prev, int s)
{
    __shared__ float sMu[D_*(OUT_+1)];
    __shared__ float sBt[D_];
    __shared__ float sY[OUT_];

    const int b = blockIdx.x, tid = threadIdx.x;
    const int dd = tid >> 3, l8 = tid & 7;        // 32 variables x 8 threads
    const float* gb = ws_g + ((size_t)b*D_ + dd)*H_;
    const float* hb = ws_h + ((size_t)b*D_ + dd)*H_;

    float acc[4];
    #pragma unroll
    for (int u = 0; u < 4; ++u) acc[u] = Phb[l8*4 + u];
    float btacc = 0.f;
    for (int k = 0; k < 2*H_; ++k){
        float hg = (k < H_) ? gb[k] : hb[k - H_];
        #pragma unroll
        for (int u = 0; u < 4; ++u) acc[u] += hg * Phw[k*OUT_ + l8*4 + u];
        btacc += hg * Fbw[k];
    }
    #pragma unroll
    for (int u = 0; u < 4; ++u) sMu[dd*(OUT_+1) + l8*4 + u] = acc[u];
    if (l8 == 0) sBt[dd] = __expf(tanhf(btacc + Fbb[0]));
    __syncthreads();

    if (tid < D_){
        float p = sBt[tid], tot = p;
        #pragma unroll
        for (int off = 16; off > 0; off >>= 1) tot += __shfl_xor(tot, off);
        float beta = p / tot;
        sBt[tid] = beta;
        out_betas[((size_t)b*FSTEPS_ + s)*D_ + tid] = beta;
    }
    __syncthreads();
    if (tid < OUT_){
        float y = 0.f;
        #pragma unroll
        for (int d2 = 0; d2 < D_; ++d2) y += sBt[d2]*sMu[d2*(OUT_+1) + tid];
        sY[tid] = y;
        out_fore[((size_t)b*FSTEPS_ + s)*OUT_ + tid] = y;
    }
    __syncthreads();
    if (tid < D_){
        float pv = Pb[tid];
        #pragma unroll
        for (int o = 0; o < OUT_; ++o) pv += sY[o]*Pw[o*D_ + tid];
        prev[(size_t)b*D_ + tid] = pv;
    }
}

// ---------------------------------------------------------------------------
// Forecast cell: one WMMA LSTM step, state in global workspace.
// ---------------------------------------------------------------------------
__global__ __launch_bounds__(256, 1) void fcell_kernel(
    const float* __restrict__ Uj, const float* __restrict__ Ui,
    const float* __restrict__ Uf, const float* __restrict__ Uo,
    const float* __restrict__ Wj, const float* __restrict__ Wi,
    const float* __restrict__ Wf, const float* __restrict__ Wo,
    const float* __restrict__ bj, const float* __restrict__ bi,
    const float* __restrict__ bf, const float* __restrict__ bo,
    const float* __restrict__ prev,
    float* __restrict__ ws_h, float* __restrict__ ws_c)
{
    __shared__ _Float16 sW[4*H_*WS_];
    __shared__ _Float16 sH[ROWS_*WS_];
    __shared__ float    sG[4*ROWS_*GS_];
    __shared__ float    sU[4*H_], sBias[4*H_], sX[ROWS_];

    const int tid  = threadIdx.x;
    const int wave = tid >> 5, lane = tid & 31;
    const int d    = blockIdx.x & (D_-1);
    const int row0 = (blockIdx.x >> 5) * ROWS_;

    const float* Wsrc[4] = {Wj, Wi, Wf, Wo};
    const float* Usrc[4] = {Uj, Ui, Uf, Uo};
    const float* bsrc[4] = {bj, bi, bf, bo};

    for (int g = 0; g < 4; ++g){
        const float* Wg = Wsrc[g] + (size_t)d*H_*H_;
        for (int idx = tid; idx < H_*H_; idx += 256){
            int k = idx >> 7, n = idx & (H_-1);
            sW[g*H_*WS_ + n*WS_ + k] = (_Float16)Wg[idx];
        }
    }
    for (int idx = tid; idx < 4*H_; idx += 256){
        int g = idx >> 7, col = idx & (H_-1);
        sU[idx]    = Usrc[g][d*H_ + col];
        sBias[idx] = bsrc[g][d*H_ + col];
    }
    for (int e = tid; e < ROWS_*H_; e += 256){
        int r = e >> 7, c = e & (H_-1);
        sH[r*WS_ + c] = (_Float16)ws_h[((size_t)(row0+r)*D_ + d)*H_ + c];
    }
    if (tid < ROWS_) sX[tid] = prev[(size_t)(row0+tid)*D_ + d];
    __syncthreads();

    #pragma unroll
    for (int q = 0; q < 8; ++q){
        const int tIdx = wave*8 + q;
        const int g = tIdx >> 4, rem = tIdx & 15, mt = rem >> 3, nt = rem & 7;
        const int hs8  = (lane >> 4) << 3;
        const int arow = mt*16 + (lane & 15);
        const int bcol = nt*16 + (lane & 15);
        v8f acc = {};
        #pragma unroll
        for (int kb = 0; kb < 4; ++kb){
            const _Float16* ap = &sH[arow*WS_ + kb*32 + hs8];
            const _Float16* bp = &sW[g*H_*WS_ + bcol*WS_ + kb*32 + hs8];
            v16h a, b;
            #pragma unroll
            for (int i = 0; i < 8; ++i){
                a[i] = ap[i];  a[i+8] = ap[16+i];
                b[i] = bp[i];  b[i+8] = bp[16+i];
            }
            acc = __builtin_amdgcn_wmma_f32_16x16x32_f16(
                      false, a, false, b, (short)0, acc, false, false);
        }
        const int cm0 = mt*16 + hs8;
        #pragma unroll
        for (int v = 0; v < 8; ++v)
            sG[(g*ROWS_ + cm0 + v)*GS_ + bcol] = acc[v];
    }
    __syncthreads();

    for (int e = tid; e < ROWS_*H_; e += 256){
        int r = e >> 7, c = e & (H_-1);
        size_t o = ((size_t)(row0+r)*D_ + d)*H_ + c;
        float xr = sX[r];
        float jv = tanhf    (sG[(0*ROWS_+r)*GS_+c] + xr*sU[0*H_+c] + sBias[0*H_+c]);
        float iv = sigmoidf_(sG[(1*ROWS_+r)*GS_+c] + xr*sU[1*H_+c] + sBias[1*H_+c]);
        float fv = sigmoidf_(sG[(2*ROWS_+r)*GS_+c] + xr*sU[2*H_+c] + sBias[2*H_+c]);
        float ov = sigmoidf_(sG[(3*ROWS_+r)*GS_+c] + xr*sU[3*H_+c] + sBias[3*H_+c]);
        float c2 = ws_c[o]*fv + iv*jv;
        float h2 = ov*tanhf(c2);
        ws_c[o] = c2;
        ws_h[o] = h2;
    }
}

// ---------------------------------------------------------------------------
extern "C" void kernel_launch(void* const* d_in, const int* in_sizes, int n_in,
                              void* d_out, int out_size, void* d_ws, size_t ws_size,
                              hipStream_t stream)
{
    (void)in_sizes; (void)n_in; (void)out_size; (void)ws_size;
    const float* x   = (const float*)d_in[0];
    const float* Uj  = (const float*)d_in[1];
    const float* Ui  = (const float*)d_in[2];
    const float* Uf  = (const float*)d_in[3];
    const float* Uo  = (const float*)d_in[4];
    const float* Wj  = (const float*)d_in[5];
    const float* Wi  = (const float*)d_in[6];
    const float* Wf  = (const float*)d_in[7];
    const float* Wo  = (const float*)d_in[8];
    const float* bj  = (const float*)d_in[9];
    const float* bi  = (const float*)d_in[10];
    const float* bf  = (const float*)d_in[11];
    const float* bo  = (const float*)d_in[12];
    const float* Fa  = (const float*)d_in[13];
    const float* Fab = (const float*)d_in[14];
    const float* Fbw = (const float*)d_in[15];
    const float* Fbb = (const float*)d_in[16];
    const float* Phw = (const float*)d_in[17];
    const float* Phb = (const float*)d_in[18];
    const float* Pw  = (const float*)d_in[19];
    const float* Pb  = (const float*)d_in[20];

    float* out       = (float*)d_out;
    float* out_fore  = out;                                   // [B,F,OUT]
    float* out_alph  = out_fore + (size_t)B_*FSTEPS_*OUT_;    // [B,T,D]
    float* out_beta  = out_alph + (size_t)B_*T_*D_;           // [B,F,D]
    float* out_outs  = out_beta + (size_t)B_*FSTEPS_*D_;      // [B,T,D,H]

    float* ws      = (float*)d_ws;
    float* ws_h    = ws;
    float* ws_c    = ws_h + (size_t)B_*D_*H_;
    float* ws_g    = ws_c + (size_t)B_*D_*H_;
    float* ws_prev = ws_g + (size_t)B_*D_*H_;

    const int cellGrid = D_ * (B_ / ROWS_);   // 256 workgroups

    enc_kernel<<<cellGrid, 256, 0, stream>>>(
        x, Uj, Ui, Uf, Uo, Wj, Wi, Wf, Wo, bj, bi, bf, bo, Fa, Fab,
        out_alph, out_outs, ws_h, ws_c, ws_g);

    for (int s = 0; s < FSTEPS_; ++s){
        fhead_kernel<<<B_, 256, 0, stream>>>(
            ws_g, ws_h, Fbw, Fbb, Phw, Phb, Pw, Pb,
            out_fore, out_beta, ws_prev, s);
        fcell_kernel<<<cellGrid, 256, 0, stream>>>(
            Uj, Ui, Uf, Uo, Wj, Wi, Wf, Wo, bj, bi, bf, bo,
            ws_prev, ws_h, ws_c);
    }
}
